// ResidualQuantizationLayer_71820443124072
// MI455X (gfx1250) — matrix operand order repaired and checked by reading, not compiled, not run on hardware
//
#include <hip/hip_runtime.h>
#include <hip/hip_bf16.h>

// Problem constants (from reference)
#define NCB    8        // codebooks
#define DIM    512      // latent
#define KCW    2048     // codewords per level
#define BATCH  16384
#define MROWS  64       // batch rows per block (4 m-tiles of 16)

// Output layout in d_out (4-byte elements, concatenated in reference return order):
//   [0 .. 131071]                 indices int32 [B,8]  (written via int* alias)
//   [131072 .. 8519679]           quantized_x float [B,512]
//   [8519680]                     num_small_clusters (float)
//   [8519681]                     sum_quant_loss     (float)
#define OUT_QX_OFF    (BATCH * NCB)
#define OUT_NS_OFF    (OUT_QX_OFF + BATCH * DIM)
#define OUT_LOSS_OFF  (OUT_NS_OFF + 1)

// Workspace: bf16 transposed embeds [l][k][d] (16 MB) then enorm fp32 [l][k] (64 KB)
#define WS_BF16_ELEMS ((size_t)NCB * KCW * DIM)

typedef __attribute__((ext_vector_type(16))) __bf16 v16bf;
typedef __attribute__((ext_vector_type(8)))  float  v8f;
typedef __attribute__((ext_vector_type(4)))  int    v4i;
typedef __attribute__((address_space(1)))    v4i    gv4i;   // global
typedef __attribute__((address_space(3)))    v4i    lv4i;   // LDS

#if __has_builtin(__builtin_amdgcn_global_load_async_to_lds_b128) && \
    __has_builtin(__builtin_amdgcn_s_wait_asynccnt)
#define ASYNC_LDS 1
#else
#define ASYNC_LDS 0
#endif

// ---------------------------------------------------------------------------
__global__ void rq_init_scalars(float* out) {
    if (threadIdx.x == 0) {
        out[OUT_NS_OFF]   = 0.0f;
        out[OUT_LOSS_OFF] = 0.0f;
    }
}

// ---------------------------------------------------------------------------
// embeds [l][d][k] fp32 -> ws [l][k][d] bf16 via 32x32 LDS tile transpose.
__global__ void rq_transpose_bf16(const float* __restrict__ embeds, __bf16* __restrict__ wsbf) {
    __shared__ float tile[32][33];
    const int l  = blockIdx.z;
    const int d0 = blockIdx.y * 32;
    const int k0 = blockIdx.x * 32;
    const float* src = embeds + (size_t)l * DIM * KCW;
    const int tid = threadIdx.x;
#pragma unroll
    for (int t = 0; t < 4; ++t) {
        int idx = tid + t * 256;
        int i = idx >> 5, j = idx & 31;
        tile[i][j] = src[(size_t)(d0 + i) * KCW + (k0 + j)];
    }
    __syncthreads();
    __bf16* dst = wsbf + (size_t)l * KCW * DIM;
#pragma unroll
    for (int t = 0; t < 4; ++t) {
        int idx = tid + t * 256;
        int kk = idx >> 5, dd = idx & 31;
        dst[(size_t)(k0 + kk) * DIM + (d0 + dd)] = (__bf16)tile[dd][kk];
    }
}

// ---------------------------------------------------------------------------
__global__ void rq_enorm(const float* __restrict__ embeds, float* __restrict__ enorm) {
    int gid = blockIdx.x * blockDim.x + threadIdx.x;
    if (gid >= NCB * KCW) return;
    int l = gid >> 11, k = gid & (KCW - 1);
    const float* src = embeds + (size_t)l * DIM * KCW + k;
    float s = 0.0f;
#pragma unroll 4
    for (int d = 0; d < DIM; ++d) s += src[(size_t)d * KCW] * src[(size_t)d * KCW];
    enorm[gid] = s;
}

// ---------------------------------------------------------------------------
__global__ void rq_small_clusters(const float* __restrict__ cs, float* out) {
    int gid = blockIdx.x * blockDim.x + threadIdx.x;
    float c = (gid < NCB * KCW && cs[gid] < 1.0f) ? 1.0f : 0.0f;
#pragma unroll
    for (int off = 1; off < 32; off <<= 1) c += __shfl_xor(c, off, 32);
    if ((threadIdx.x & 31) == 0 && c != 0.0f) atomicAdd(&out[OUT_NS_OFF], c);
}

// ---------------------------------------------------------------------------
// Stage one 32-codeword group (32 KB bf16) into LDS, 256 threads x 8 chunks of 16B.
__device__ __forceinline__ void rq_stage(const __bf16* __restrict__ lw, int group,
                                         __bf16* dst, int tid) {
    const __bf16* gsrc = lw + (size_t)group * 32 * DIM;
#if ASYNC_LDS
#pragma unroll
    for (int t = 0; t < 8; ++t) {
        int c = tid + t * 256;   // 16-byte chunk id, 0..2047
        v4i* gp = (v4i*)(gsrc + c * 8);   // strips const, generic
        v4i* lp = (v4i*)(dst + c * 8);
        __builtin_amdgcn_global_load_async_to_lds_b128((gv4i*)gp, (lv4i*)lp, 0, 0);
    }
#else
#pragma unroll
    for (int t = 0; t < 8; ++t) {
        int c = tid + t * 256;
        *(v16bf*)(dst + c * 8) = __builtin_nontemporal_load((const v16bf*)(gsrc + c * 8));
    }
#endif
}

__device__ __forceinline__ void rq_stage_wait_prev() {
#if ASYNC_LDS
    __builtin_amdgcn_s_wait_asynccnt(8);   // oldest staged group (8 instr/wave) retired
#endif
}
__device__ __forceinline__ void rq_stage_wait_all() {
#if ASYNC_LDS
    __builtin_amdgcn_s_wait_asynccnt(0);
#endif
}

// ---------------------------------------------------------------------------
// Main kernel: one block = 64 batch rows, 256 threads = 8 waves.
// Wave w -> m-tile (w&3), n-parity (w>>2). B groups of 32 codewords are staged
// into LDS (async, double-buffered) and shared by all 8 waves.
__global__ __launch_bounds__(256)
void rq_main(const float* __restrict__ x,
             const __bf16* __restrict__ wsbf,
             const float* __restrict__ enorm,
             int*   __restrict__ outIdx,
             float* __restrict__ outQx,
             float* __restrict__ outScal) {
    __shared__ float  resLDS[MROWS][DIM];        // 128 KB fp32 residual tile
    __shared__ __bf16 ldsB[2][32][DIM];          // 64 KB double-buffered B groups
    __shared__ float  wbScore[2][MROWS];
    __shared__ int    wbIdx[2][MROWS];
    __shared__ int    bestIdxLDS[MROWS];

    const int tid  = threadIdx.x;
    const int w    = tid >> 5;
    const int lane = tid & 31;
    const int mt   = w & 3;                      // m-tile 0..3
    const int par  = w >> 2;                     // n-tile parity 0/1
    const size_t rowBase = (size_t)blockIdx.x * MROWS;

    for (int e = tid; e < MROWS * DIM; e += 256) {
        int m = e >> 9, d = e & (DIM - 1);
        resLDS[m][d] = x[(rowBase + m) * DIM + d];
    }
    __syncthreads();

    const int nlocal = lane & 15;
    const int abase  = (lane < 16) ? 0 : 8;      // A-fragment K group (contiguous order k = 16h + s, via 2*abase)
    const int koff   = (lane < 16) ? 0 : 16;     // B-fragment K offset (contiguous order k = 16h + s, same as A)

    for (int lvl = 0; lvl < NCB; ++lvl) {
        const __bf16* lw = wsbf + (size_t)lvl * KCW * DIM;
        const float*  en = enorm + (size_t)lvl * KCW;

        // Build this wave's 16 A fragments (rows mt*16 .. mt*16+15) once per level.
        v16bf afrag[16];
#pragma unroll
        for (int ks = 0; ks < 16; ++ks) {
            const float* rrow = &resLDS[mt * 16 + nlocal][ks * 32];
            v16bf a;
#pragma unroll
            for (int i = 0; i < 8; ++i) {
                a[i]     = (__bf16)rrow[2 * abase + i];
                a[i + 8] = (__bf16)rrow[2 * abase + 8 + i];
            }
            afrag[ks] = a;
        }

        float bs[8];
        int   bi[8];
#pragma unroll
        for (int r = 0; r < 8; ++r) { bs[r] = 3.0e38f; bi[r] = 0; }

        // Prime the double buffer with groups 0 and 1.
        rq_stage(lw, 0, &ldsB[0][0][0], tid);
        rq_stage(lw, 1, &ldsB[1][0][0], tid);

        for (int g = 0; g < 64; ++g) {           // 64 groups of 32 codewords
            if (g < 63) rq_stage_wait_prev();    // group g staged (one still in flight)
            else        rq_stage_wait_all();
            __syncthreads();                     // all waves' staging of group g done

            const int n = g * 32 + par * 16 + nlocal;          // this lane's codeword
            const __bf16* bp = &ldsB[g & 1][par * 16 + nlocal][koff];

            v8f c0 = {}, c1 = {};
#pragma unroll
            for (int ks = 0; ks < 16; ks += 2) { // two chains to shorten WMMA RAW depth
                v16bf b0 = *(const v16bf*)(bp + ks * 32);
                v16bf b1 = *(const v16bf*)(bp + ks * 32 + 32);
                c0 = __builtin_amdgcn_wmma_f32_16x16x32_bf16(
                         false, afrag[ks],     false, b0, (short)0, c0, false, false);
                c1 = __builtin_amdgcn_wmma_f32_16x16x32_bf16(
                         false, afrag[ks + 1], false, b1, (short)0, c1, false, false);
            }
            const float enn = en[n];
#pragma unroll
            for (int r = 0; r < 8; ++r) {
                float s = enn - 2.0f * (c0[r] + c1[r]);
                if (s < bs[r]) { bs[r] = s; bi[r] = n; }
            }

            __syncthreads();                     // everyone done reading buffer g&1
            if (g + 2 < 64) rq_stage(lw, g + 2, &ldsB[g & 1][0][0], tid);
        }

        // Cross-lane argmin within 16-lane halves (C layout: lane = N, M = r + 8*(lane>=16)).
#pragma unroll
        for (int off = 1; off < 16; off <<= 1) {
#pragma unroll
            for (int r = 0; r < 8; ++r) {
                float os = __shfl_xor(bs[r], off, 32);
                int   oi = __shfl_xor(bi[r], off, 32);
                if (os < bs[r] || (os == bs[r] && oi < bi[r])) { bs[r] = os; bi[r] = oi; }
            }
        }
        if (nlocal == 0) {
            int mrow = mt * 16 + ((lane < 16) ? 0 : 8);
#pragma unroll
            for (int r = 0; r < 8; ++r) {
                wbScore[par][mrow + r] = bs[r];
                wbIdx[par][mrow + r]   = bi[r];
            }
        }
        __syncthreads();

        // Merge the two n-parities, emit index output.
        if (tid < MROWS) {
            float s0 = wbScore[0][tid], s1 = wbScore[1][tid];
            int   i0 = wbIdx[0][tid],   i1 = wbIdx[1][tid];
            int bidx = (s1 < s0 || (s1 == s0 && i1 < i0)) ? i1 : i0;
            bestIdxLDS[tid] = bidx;
            outIdx[(rowBase + tid) * NCB + lvl] = bidx;
        }
        __syncthreads();

        // Residual update + loss: loss_l = mean((quant-res)^2) = mean(new_res^2).
        float lsum = 0.0f;
        for (int e = tid; e < MROWS * DIM; e += 256) {
            int m = e >> 9, d = e & (DIM - 1);
            float q  = (float)lw[(size_t)bestIdxLDS[m] * DIM + d];
            float nr = resLDS[m][d] - q;
            resLDS[m][d] = nr;
            lsum += nr * nr;
        }
#pragma unroll
        for (int off = 1; off < 32; off <<= 1) lsum += __shfl_xor(lsum, off, 32);
        if (lane == 0)
            atomicAdd(&outScal[OUT_LOSS_OFF], lsum * (1.0f / ((float)BATCH * (float)DIM)));
        __syncthreads();
    }

    // quantized_x = x - final residual
    for (int e = tid; e < MROWS * DIM; e += 256) {
        int m = e >> 9, d = e & (DIM - 1);
        size_t gph = (rowBase + m) * DIM + d;
        outQx[gph] = x[gph] - resLDS[m][d];
    }
}

// ---------------------------------------------------------------------------
extern "C" void kernel_launch(void* const* d_in, const int* in_sizes, int n_in,
                              void* d_out, int out_size, void* d_ws, size_t ws_size,
                              hipStream_t stream) {
    const float* x      = (const float*)d_in[0];   // [16384, 512]
    const float* embeds = (const float*)d_in[1];   // [8, 512, 2048]
    const float* cs     = (const float*)d_in[2];   // [8, 2048]

    float* out   = (float*)d_out;
    int*   outI  = (int*)d_out;
    float* outQx = out + OUT_QX_OFF;

    __bf16* wsbf  = (__bf16*)d_ws;
    float*  enorm = (float*)((char*)d_ws + WS_BF16_ELEMS * 2);

    rq_init_scalars<<<1, 64, 0, stream>>>(out);
    {
        dim3 g(KCW / 32, DIM / 32, NCB);
        rq_transpose_bf16<<<g, 256, 0, stream>>>(embeds, wsbf);
    }
    rq_enorm<<<(NCB * KCW) / 256, 256, 0, stream>>>(embeds, enorm);
    rq_small_clusters<<<(NCB * KCW) / 256, 256, 0, stream>>>(cs, out);

    rq_main<<<BATCH / MROWS, 256, 0, stream>>>(x, wsbf, enorm, outI, outQx, out);
}
